// QLinearQAT_86552180949217
// MI455X (gfx1250) — 8-bit matrix operand order repaired and checked by reading, compiled, not run on hardware
//
#include <hip/hip_runtime.h>
#include <hip/hip_bf16.h>
#include <stdint.h>

#define TOKENS 8192
#define D_IN   4096
#define D_OUT  4096

#define BLK_M 256
#define BLK_N 128
#define KSTEP 64

typedef __attribute__((ext_vector_type(8))) int      v8i;
typedef __attribute__((ext_vector_type(4))) int      v4i;
typedef __attribute__((ext_vector_type(4))) unsigned v4u;

#define WS_X8_OFF 256
#define WS_W8_OFF (256 + (size_t)TOKENS * (size_t)D_IN)

#if __has_builtin(__builtin_amdgcn_tensor_load_to_lds) && __has_builtin(__builtin_amdgcn_s_wait_tensorcnt)
#define USE_TDM 1
#else
#define USE_TDM 0
#endif

__global__ void qlin_init_ws(unsigned* cv) {
    if (threadIdx.x == 0 && blockIdx.x == 0) cv[0] = 0u;
}

__global__ void qlin_reduce_max(const float4* __restrict__ x, unsigned* __restrict__ cv, int n4) {
    float m = 0.0f;
    int stride = gridDim.x * blockDim.x;
    for (int i = blockIdx.x * blockDim.x + threadIdx.x; i < n4; i += stride) {
        float4 v = x[i];
        m = fmaxf(m, fmaxf(fmaxf(v.x, v.y), fmaxf(v.z, v.w)));
    }
    for (int off = 16; off > 0; off >>= 1)
        m = fmaxf(m, __shfl_xor(m, off, 32));
    __shared__ float smax[8];
    int lane = threadIdx.x & 31;
    int wave = threadIdx.x >> 5;
    if (lane == 0) smax[wave] = m;
    __syncthreads();
    if (threadIdx.x == 0) {
        float bm = smax[0];
        #pragma unroll
        for (int i = 1; i < 8; ++i) bm = fmaxf(bm, smax[i]);
        atomicMax(cv, __float_as_uint(fmaxf(bm, 0.0f)));
    }
}

__global__ void qlin_quant_x(const float4* __restrict__ x, const float* __restrict__ cv,
                             char4* __restrict__ x8, int n4) {
    const float r = 127.0f / cv[0];
    int stride = gridDim.x * blockDim.x;
    for (int i = blockIdx.x * blockDim.x + threadIdx.x; i < n4; i += stride) {
        float4 v = x[i];
        float a = fminf(fmaxf(rintf(v.x * r), -128.0f), 127.0f);
        float b = fminf(fmaxf(rintf(v.y * r), -128.0f), 127.0f);
        float c = fminf(fmaxf(rintf(v.z * r), -128.0f), 127.0f);
        float d = fminf(fmaxf(rintf(v.w * r), -128.0f), 127.0f);
        char4 o;
        o.x = (signed char)(int)a; o.y = (signed char)(int)b;
        o.z = (signed char)(int)c; o.w = (signed char)(int)d;
        x8[i] = o;
    }
}

__global__ void qlin_quant_w(const float4* __restrict__ w, const float* __restrict__ qs,
                             char4* __restrict__ w8, int n4) {
    int stride = gridDim.x * blockDim.x;
    for (int i = blockIdx.x * blockDim.x + threadIdx.x; i < n4; i += stride) {
        int row = i >> 10;
        float rinv = 1.0f / qs[row];
        float4 v = w[i];
        float a = fminf(fmaxf(rintf(v.x * rinv), -128.0f), 127.0f);
        float b = fminf(fmaxf(rintf(v.y * rinv), -128.0f), 127.0f);
        float c = fminf(fmaxf(rintf(v.z * rinv), -128.0f), 127.0f);
        float d = fminf(fmaxf(rintf(v.w * rinv), -128.0f), 127.0f);
        char4 o;
        o.x = (signed char)(int)a; o.y = (signed char)(int)b;
        o.z = (signed char)(int)c; o.w = (signed char)(int)d;
        w8[i] = o;
    }
}

#if USE_TDM
__device__ __forceinline__ void tdm_load_tile_2d(const void* gsrc, unsigned lds_byte_off,
                                                 unsigned tensor_d0, unsigned tensor_d1,
                                                 unsigned tile_d0, unsigned tile_d1,
                                                 unsigned row_stride_bytes) {
    unsigned long long ga = (unsigned long long)(uintptr_t)gsrc;
    v4u g0;
    g0[0] = 1u;
    g0[1] = lds_byte_off;
    g0[2] = (unsigned)ga;
    g0[3] = (unsigned)((ga >> 32) & 0x01FFFFFFu)
          | (2u << 30);
    v8i g1;
    g1[0] = 0;
    g1[1] = (int)((tensor_d0 & 0xFFFFu) << 16);
    g1[2] = (int)(((tensor_d0 >> 16) & 0xFFFFu)
          |       ((tensor_d1 & 0xFFFFu) << 16));
    g1[3] = (int)(((tensor_d1 >> 16) & 0xFFFFu)
          |       ((tile_d0 & 0xFFFFu) << 16));
    g1[4] = (int)(tile_d1 & 0xFFFFu);
    g1[5] = (int)row_stride_bytes;
    g1[6] = 0;
    g1[7] = 0;
    v4i z4 = {0, 0, 0, 0};
    v8i z8 = {0, 0, 0, 0, 0, 0, 0, 0};
    __builtin_amdgcn_tensor_load_to_lds(g0, g1, z4, z4, z8, 0);
}
#endif

__global__ __launch_bounds__(256)
void qlin_gemm_i8_wmma(const char* __restrict__ x8, const char* __restrict__ w8,
                       const float* __restrict__ cv, const float* __restrict__ qscales,
                       const float* __restrict__ bias, float* __restrict__ out) {
    const int tid  = threadIdx.x;
    const int lane = tid & 31;
    const int wave = tid >> 5;
    const int blockRow = blockIdx.y * BLK_M;
    const int blockCol = blockIdx.x * BLK_N;

    const int waveRow = (wave >> 1) * 64;
    const int waveCol = (wave & 1) * 64;
    const int half = lane >> 4;
    const int lr   = lane & 15;

    v8i acc[4][4];
    #pragma unroll
    for (int rt = 0; rt < 4; ++rt)
        #pragma unroll
        for (int ct = 0; ct < 4; ++ct)
            acc[rt][ct] = (v8i){0, 0, 0, 0, 0, 0, 0, 0};

#if USE_TDM
    __shared__ __align__(16) char sA[2][BLK_M * KSTEP];
    __shared__ __align__(16) char sB[2][BLK_N * KSTEP];

    const char* gA = x8 + (size_t)blockRow * D_IN;
    const char* gB = w8 + (size_t)blockCol * D_IN;

    if (tid < 32) {
        tdm_load_tile_2d(gA, (unsigned)(size_t)&sA[0][0], D_IN, BLK_M, KSTEP, BLK_M, D_IN);
        tdm_load_tile_2d(gB, (unsigned)(size_t)&sB[0][0], D_IN, BLK_N, KSTEP, BLK_N, D_IN);
        __builtin_amdgcn_s_wait_tensorcnt(0);
    }
    __syncthreads();

    int cur = 0;
    for (int k = 0; k < D_IN; k += KSTEP) {
        const int nxt = cur ^ 1;
        if (k + KSTEP < D_IN && tid < 32) {
            tdm_load_tile_2d(gA + k + KSTEP, (unsigned)(size_t)&sA[nxt][0], D_IN, BLK_M, KSTEP, BLK_M, D_IN);
            tdm_load_tile_2d(gB + k + KSTEP, (unsigned)(size_t)&sB[nxt][0], D_IN, BLK_N, KSTEP, BLK_N, D_IN);
        }
        const char* bA = sA[cur];
        const char* bB = sB[cur];
#else
    __shared__ __align__(16) char sA[1][BLK_M * KSTEP];
    __shared__ __align__(16) char sB[1][BLK_N * KSTEP];
    for (int k = 0; k < D_IN; k += KSTEP) {
        #pragma unroll
        for (int j = 0; j < 4; ++j) {
            int c = tid + j * 256, r = c >> 2, o = (c & 3) * 16;
            *(int4*)(sA[0] + r * KSTEP + o) =
                *(const int4*)(x8 + (size_t)(blockRow + r) * D_IN + k + o);
        }
        #pragma unroll
        for (int j = 0; j < 2; ++j) {
            int c = tid + j * 256, r = c >> 2, o = (c & 3) * 16;
            *(int4*)(sB[0] + r * KSTEP + o) =
                *(const int4*)(w8 + (size_t)(blockCol + r) * D_IN + k + o);
        }
        if (k + KSTEP < D_IN) {
            __builtin_prefetch(x8 + (size_t)(blockRow + (tid >> 2)) * D_IN + k + KSTEP, 0, 3);
            __builtin_prefetch(w8 + (size_t)(blockCol + (tid >> 2)) * D_IN + k + KSTEP, 0, 3);
        }
        __syncthreads();
        const char* bA = sA[0];
        const char* bB = sB[0];
#endif
        v8i afrag[4];
        #pragma unroll
        for (int rt = 0; rt < 4; ++rt) {
            const char* p = bA + (waveRow + rt * 16 + lr) * KSTEP + half * 8;
            union { v8i v; long d[4]; } u;
            u.d[0] = *(const long*)(p);
            u.d[1] = *(const long*)(p + 16);
            u.d[2] = *(const long*)(p + 32);
            u.d[3] = *(const long*)(p + 48);
            afrag[rt] = u.v;
        }
        v8i bfrag[4];
        #pragma unroll
        for (int ct = 0; ct < 4; ++ct) {
            const char* p = bB + (waveCol + ct * 16 + lr) * KSTEP + half * 8;
            union { v8i v; long d[4]; } u;
            u.d[0] = *(const long*)(p);
            u.d[1] = *(const long*)(p + 16);
            u.d[2] = *(const long*)(p + 32);
            u.d[3] = *(const long*)(p + 48);
            bfrag[ct] = u.v;
        }

        #pragma unroll
        for (int rt = 0; rt < 4; ++rt)
            #pragma unroll
            for (int ct = 0; ct < 4; ++ct)
                acc[rt][ct] = __builtin_amdgcn_wmma_i32_16x16x64_iu8(
                     true, afrag[rt],
                     true, bfrag[ct],
                    acc[rt][ct],
                     false,  false);

#if USE_TDM
        if (tid < 32) __builtin_amdgcn_s_wait_tensorcnt(0);
        __syncthreads();
        cur = nxt;
    }
#else
        __syncthreads();
    }
#endif

    const float s = cv[0] * (1.0f / 127.0f);
    #pragma unroll
    for (int ct = 0; ct < 4; ++ct) {
        const int outch = blockCol + waveCol + ct * 16 + lr;
        const float scale = s * qscales[outch];
        const float bv = bias[outch];
        #pragma unroll
        for (int rt = 0; rt < 4; ++rt) {
            const int token0 = blockRow + waveRow + rt * 16 + half * 8;
            float* o = out + (size_t)token0 * D_OUT + outch;
            #pragma unroll
            for (int r = 0; r < 8; ++r)
                o[(size_t)r * D_OUT] = (float)acc[rt][ct][r] * scale + bv;
        }
    }
}

extern "C" void kernel_launch(void* const* d_in, const int* in_sizes, int n_in,
                              void* d_out, int out_size, void* d_ws, size_t ws_size,
                              hipStream_t stream) {
    const float* x       = (const float*)d_in[0];
    const float* weight  = (const float*)d_in[1];
    const float* qscales = (const float*)d_in[2];
    const float* bias    = (const float*)d_in[3];
    float* out = (float*)d_out;

    unsigned* cv_u = (unsigned*)d_ws;
    const float* cv_f = (const float*)d_ws;
    char* x8 = (char*)d_ws + WS_X8_OFF;
    char* w8 = (char*)d_ws + WS_W8_OFF;

    const int nX4 = TOKENS * D_IN / 4;
    const int nW4 = D_OUT * D_IN / 4;

    qlin_init_ws<<<1, 32, 0, stream>>>(cv_u);
    qlin_reduce_max<<<1024, 256, 0, stream>>>((const float4*)x, cv_u, nX4);
    qlin_quant_x<<<1024, 256, 0, stream>>>((const float4*)x, cv_f, (char4*)x8, nX4);
    qlin_quant_w<<<1024, 256, 0, stream>>>((const float4*)weight, qscales, (char4*)w8, nW4);

    dim3 grid(D_OUT / BLK_N, TOKENS / BLK_M);
    qlin_gemm_i8_wmma<<<grid, 256, 0, stream>>>(x8, w8, cv_f, qscales, bias, out);
}
